// PAiNNMMEquivariant_7370163880194
// MI455X (gfx1250) — matrix operand order repaired and checked by reading, not compiled, not run on hardware
//
#include <hip/hip_runtime.h>

// ---------------------------------------------------------------------------
// PaiNN-MM equivariant network for MI455X (gfx1250, wave32, WMMA f16)
// B=16, N=64, NM=16384, CS=CV=128, ER=32, NL=3
// ---------------------------------------------------------------------------

#define Bb 16
#define Nn 64
#define NM 16384
#define CSc 128
#define CVc 128
#define ERr 32
#define RCf 5.0f
#define PIf 3.14159265358979f
#define FACTORf 14.3996538f
#define EPSf 1e-12f
#define NODES (Bb * Nn)          // 1024
#define PAIRS (Bb * Nn * Nn)     // 65536
#define K1P 320                  // 291 padded to 320 (10 k-steps of 32)

typedef __attribute__((ext_vector_type(16))) _Float16 v16h;
typedef __attribute__((ext_vector_type(8)))  _Float16 v8h;
typedef __attribute__((ext_vector_type(8)))  float    v8f;

__device__ __forceinline__ float silu_f(float x) { return x / (1.0f + __expf(-x)); }

// ---------------------------------------------------------------------------
// Weight packing: fp32 [K,N] row-major -> f16 B-fragment layout.
// Tile (kt,nt); lane l holds column n = nt*16 + (l&15), rows
// k = kt*32 + ((l&16)?8:0) + (e<8 ? e : 16+e-8), e=0..15 (table order).
// ---------------------------------------------------------------------------
__global__ void pack_w_kernel(const float* __restrict__ W, _Float16* __restrict__ dst,
                              int K, int N, int Kp) {
    int total = Kp * N;
    int Nt = N >> 4;
    for (int idx = blockIdx.x * blockDim.x + threadIdx.x; idx < total;
         idx += gridDim.x * blockDim.x) {
        int e    = idx & 15;
        int lane = (idx >> 4) & 31;
        int tl   = idx >> 9;
        int kt   = tl / Nt;
        int nt   = tl - kt * Nt;
        int k = kt * 32 + ((lane & 16) ? 8 : 0) + ((e < 8) ? e : (16 + e - 8));
        int n = nt * 16 + (lane & 15);
        dst[idx] = (_Float16)((k < K) ? W[k * N + n] : 0.0f);
    }
}

// ---------------------------------------------------------------------------
// ESP / E-field over MM atoms. One block per QM node, reduce over NM.
// ---------------------------------------------------------------------------
__global__ void esp_kernel(const float* __restrict__ qm, const float* __restrict__ mm,
                           const float* __restrict__ q, const int* __restrict__ realmm,
                           float* __restrict__ phi, float* __restrict__ evec,
                           float* __restrict__ enorm) {
    int node = blockIdx.x;
    int b = node >> 6;
    float qx = qm[node * 3 + 0], qy = qm[node * 3 + 1], qz = qm[node * 3 + 2];
    float ap = 0.f, ax = 0.f, ay = 0.f, az = 0.f;
    for (int m = threadIdx.x; m < NM; m += blockDim.x) {
        int mi = b * NM + m;
        float rx = qx - mm[mi * 3 + 0];
        float ry = qy - mm[mi * 3 + 1];
        float rz = qz - mm[mi * 3 + 2];
        float d = sqrtf(rx * rx + ry * ry + rz * rz);
        float dm = fmaxf(d, EPSf);
        float msk = realmm[mi] ? 1.0f : 0.0f;
        float invd = msk / dm;
        float invd3 = invd / (dm * dm);
        float qv = q[mi];
        ap += qv * invd;
        ax += qv * rx * invd3;
        ay += qv * ry * invd3;
        az += qv * rz * invd3;
    }
    __shared__ float red[256];
    float vals[4] = {ap, ax, ay, az};
    float out[4];
#pragma unroll
    for (int s = 0; s < 4; ++s) {
        red[threadIdx.x] = vals[s];
        __syncthreads();
        for (int w = 128; w > 0; w >>= 1) {
            if ((int)threadIdx.x < w) red[threadIdx.x] += red[threadIdx.x + w];
            __syncthreads();
        }
        out[s] = red[0];
        __syncthreads();
    }
    if (threadIdx.x == 0) {
        float ex = out[1] * FACTORf, ey = out[2] * FACTORf, ez = out[3] * FACTORf;
        phi[node] = out[0] * FACTORf;
        evec[node * 3 + 0] = ex;
        evec[node * 3 + 1] = ey;
        evec[node * 3 + 2] = ez;
        enorm[node] = fmaxf(sqrtf(ex * ex + ey * ey + ez * ez), EPSf);
    }
}

// ---------------------------------------------------------------------------
// Node initialization: phi_mlp, s_proj, E_gate. One 128-thread block per node.
// ---------------------------------------------------------------------------
__global__ void init_nodes_kernel(
    const float* __restrict__ phi, const float* __restrict__ enorm,
    const float* __restrict__ evec, const int* __restrict__ atype,
    const float* __restrict__ embed,
    const float* pW1, const float* pB1, const float* pW2, const float* pB2,
    const float* pW3, const float* pB3,
    const float* spW, const float* spB,
    const float* eW1, const float* eB1, const float* eW2, const float* eB2,
    const float* eW3, const float* eB3,
    float* __restrict__ h_s, float* __restrict__ h_v) {
    int node = blockIdx.x;
    int i = node & 63;
    int t = threadIdx.x;
    __shared__ float h1[64], h2[64], in_s[161], hs[128], g1[64], g2[64];
    float x0 = phi[node];
    float En = enorm[node];

    if (t < 64) h1[t] = silu_f(pW1[t] * x0 + pB1[t]);  // W(1,64)
    __syncthreads();
    if (t < 64) {
        float a = pB2[t];
        for (int k = 0; k < 64; ++k) a += h1[k] * pW2[k * 64 + t];
        h2[t] = silu_f(a);
    }
    __syncthreads();
    if (t < 32) {
        float a = pB3[t];
        for (int k = 0; k < 64; ++k) a += h2[k] * pW3[k * 32 + t];
        in_s[128 + t] = a;  // phi_feat
    }
    if (t < 128) in_s[t] = embed[atype[i] * CSc + t];
    if (t == 0) in_s[160] = En;
    __syncthreads();
    if (t < 128) {
        float a = spB[t];
        for (int k = 0; k < 161; ++k) a += in_s[k] * spW[k * 128 + t];
        hs[t] = a;
        h_s[node * CSc + t] = a;
    }
    __syncthreads();
    if (t < 64) {
        float a = eB1[t] + En * eW1[128 * 64 + t];
        for (int k = 0; k < 128; ++k) a += hs[k] * eW1[k * 64 + t];
        g1[t] = silu_f(a);
    }
    __syncthreads();
    if (t < 64) {
        float a = eB2[t];
        for (int k = 0; k < 64; ++k) a += g1[k] * eW2[k * 64 + t];
        g2[t] = silu_f(a);
    }
    __syncthreads();
    if (t < 128) {
        float a = eB3[t];
        for (int k = 0; k < 64; ++k) a += g2[k] * eW3[k * 128 + t];
        float gam = tanhf(a);
        float ex = evec[node * 3 + 0], ey = evec[node * 3 + 1], ez = evec[node * 3 + 2];
        h_v[(node * 3 + 0) * CVc + t] = gam * ex;
        h_v[(node * 3 + 1) * CVc + t] = gam * ey;
        h_v[(node * 3 + 2) * CVc + t] = gam * ez;
    }
}

// ---------------------------------------------------------------------------
// Pair geometry: rhat, bessel*cutoff edge features (f16), mask.
// ---------------------------------------------------------------------------
__global__ void geometry_kernel(const float* __restrict__ qm, float* __restrict__ rhat,
                                _Float16* __restrict__ eij, float* __restrict__ maskf) {
    int p = blockIdx.x * blockDim.x + threadIdx.x;
    if (p >= PAIRS) return;
    int b = p >> 12;
    int i = (p >> 6) & 63;
    int j = p & 63;
    float rx = qm[(b * Nn + i) * 3 + 0] - qm[(b * Nn + j) * 3 + 0];
    float ry = qm[(b * Nn + i) * 3 + 1] - qm[(b * Nn + j) * 3 + 1];
    float rz = qm[(b * Nn + i) * 3 + 2] - qm[(b * Nn + j) * 3 + 2];
    float d2 = fmaxf(rx * rx + ry * ry + rz * rz, EPSf * EPSf);
    float d = sqrtf(d2);
    float inv = 1.0f / d;
    rhat[p * 3 + 0] = rx * inv;
    rhat[p * 3 + 1] = ry * inv;
    rhat[p * 3 + 2] = rz * inv;
    float inRC = (d <= RCf) ? 1.0f : 0.0f;
    maskf[p] = (i != j) ? inRC : 0.0f;
    float xc = fminf(fmaxf(d / RCf, 0.0f), 1.0f);
    float fc = 0.5f * (__cosf(PIf * xc) + 1.0f) * inRC;
    const float s2rc = 0.6324555320f;  // sqrt(2/RC)
#pragma unroll 4
    for (int n = 1; n <= ERr; ++n) {
        float arg = d * (float)n * PIf / RCf;
        float v = (d > EPSf) ? (__sinf(arg) * inv) : ((float)n * PIf / RCf);
        eij[p * ERr + (n - 1)] = (_Float16)(v * s2rc * fc);
    }
}

// ---------------------------------------------------------------------------
// Per-layer pair invariant scalars c1,c2,c3. Block = (b,i), thread = j.
// h_v layout: [node][d][c] flat.
// ---------------------------------------------------------------------------
__global__ void pair_scalars_kernel(const float* __restrict__ h_v,
                                    const float* __restrict__ rhat,
                                    float* __restrict__ c1, float* __restrict__ c2,
                                    float* __restrict__ c3) {
    int bi = blockIdx.x;
    int b = bi >> 6;
    __shared__ float hvi[384];
    for (int k = threadIdx.x; k < 384; k += 64) hvi[k] = h_v[bi * 384 + k];
    __syncthreads();
    int j = threadIdx.x;
    const float* hvj = h_v + (b * Nn + j) * 384;
    float dotv = 0.f, svj0 = 0.f, svj1 = 0.f, svj2 = 0.f;
    float svi0 = 0.f, svi1 = 0.f, svi2 = 0.f;
    for (int c = 0; c < 128; ++c) {
        float a0 = hvj[c], a1 = hvj[128 + c], a2 = hvj[256 + c];
        float i0 = hvi[c], i1 = hvi[128 + c], i2 = hvi[256 + c];
        dotv += i0 * a0 + i1 * a1 + i2 * a2;
        svj0 += a0; svj1 += a1; svj2 += a2;
        svi0 += i0; svi1 += i1; svi2 += i2;
    }
    int p = bi * Nn + j;
    float r0 = rhat[p * 3 + 0], r1 = rhat[p * 3 + 1], r2 = rhat[p * 3 + 2];
    c1[p] = dotv;
    c2[p] = svi0 * r0 + svi1 * r1 + svi2 * r2;
    c3[p] = svj0 * r0 + svj1 * r1 + svj2 * r2;
}

// ---------------------------------------------------------------------------
// WMMA helpers (16x16x32 f16, f32 accumulate)
// ---------------------------------------------------------------------------
__device__ __forceinline__ v8f wmma_tile(const _Float16* sA, int ldA, int Kt, int mt,
                                         const _Float16* Wp, int Nt, int nt,
                                         unsigned lane) {
    v8f acc = {};
    int m = mt * 16 + (lane & 15);
    int ksel = (lane & 16) ? 8 : 0;
    const _Float16* arow = sA + m * ldA + ksel;
    const _Float16* wp = Wp + ((size_t)nt * 32 + lane) * 16;
    for (int kt = 0; kt < Kt; ++kt) {
        v8h lo = *(const v8h*)(arow + kt * 32);
        v8h hi = *(const v8h*)(arow + kt * 32 + 16);
        v16h a;
#pragma unroll
        for (int t = 0; t < 8; ++t) { a[t] = lo[t]; a[t + 8] = hi[t]; }
        v16h bf = *(const v16h*)(wp + (size_t)kt * Nt * 32 * 16);
        acc = __builtin_amdgcn_wmma_f32_16x16x32_f16(false, a, false, bf, (short)0, acc,
                                                     false, false);
    }
    return acc;
}

__device__ __forceinline__ void gemm_silu(const _Float16* sA, int ldA, int Kt,
                                          const _Float16* Wp, const float* bias, int Nt,
                                          _Float16* sOut, int ldO, unsigned lane,
                                          int wave) {
    int tiles = 2 * Nt;  // 32 rows = 2 M-tiles
    for (int tile = wave; tile < tiles; tile += 8) {
        int mt = tile / Nt;
        int nt = tile - mt * Nt;
        v8f acc = wmma_tile(sA, ldA, Kt, mt, Wp, Nt, nt, lane);
        int n = nt * 16 + (lane & 15);
        int mb = mt * 16 + ((lane & 16) ? 8 : 0);
        float bv = bias[n];
#pragma unroll
        for (int v = 0; v < 8; ++v) {
            float x = acc[v] + bv;
            sOut[(mb + v) * ldO + n] = (_Float16)silu_f(x);
        }
    }
}

// ---------------------------------------------------------------------------
// Fused pair MLP (phi_s + phi_ab) + masked reductions into M_s / M_v.
// Block = one (b,i) and 32 j rows. 256 threads = 8 wave32.
// ---------------------------------------------------------------------------
__global__ __launch_bounds__(256) void pair_mlp_kernel(
    const float* __restrict__ h_s, const _Float16* __restrict__ eij,
    const float* __restrict__ c1, const float* __restrict__ c2,
    const float* __restrict__ c3, const float* __restrict__ maskf,
    const float* __restrict__ rhat, const float* __restrict__ h_v,
    const _Float16* Ws1, const _Float16* Ws2, const _Float16* Ws3,
    const _Float16* Wab1, const _Float16* Wab2, const _Float16* Wab3,
    const float* bs1, const float* bs2, const float* bs3,
    const float* bab1, const float* bab2, const float* bab3,
    float* __restrict__ M_s, float* __restrict__ M_v) {
    __shared__ alignas(32) unsigned char smem[57344];
    _Float16* sX   = (_Float16*)(smem);           // 32x320 f16  (20480 B)
    _Float16* H1s  = (_Float16*)(smem + 20480);   // 32x128 f16
    _Float16* H1ab = (_Float16*)(smem + 28672);
    _Float16* H2s  = (_Float16*)(smem + 36864);
    _Float16* H2ab = (_Float16*)(smem + 45056);
    float* outS  = (float*)(smem);                // 32x128 f32 (reuses sX)
    float* outAB = (float*)(smem + 16384);        // 32x256 f32 (reuses H1ab/H2*)
    float* mskS  = (float*)(smem + 53248);        // 32 floats

    int blk = blockIdx.x;
    int bi = blk >> 1;
    int j0 = (blk & 1) * 32;
    int b = bi >> 6;
    int tid = threadIdx.x;
    unsigned lane = tid & 31;
    int wave = tid >> 5;
    int pbase = bi * Nn + j0;

    // Build 32x320 f16 feature tile: [h_s(i)|h_s(j)|eij|c1|c2|c3|pad]
    const float* hsi = h_s + bi * CSc;
    for (int idx = tid; idx < 32 * K1P; idx += 256) {
        int r = idx / K1P;
        int k = idx - r * K1P;
        float v;
        if (k < 128)      v = hsi[k];
        else if (k < 256) v = h_s[(b * Nn + j0 + r) * CSc + (k - 128)];
        else if (k < 288) v = (float)eij[(pbase + r) * ERr + (k - 256)];
        else if (k == 288) v = c1[pbase + r];
        else if (k == 289) v = c2[pbase + r];
        else if (k == 290) v = c3[pbase + r];
        else v = 0.0f;
        sX[r * K1P + k] = (_Float16)v;
    }
    if (tid < 32) mskS[tid] = maskf[pbase + tid];
    __syncthreads();

    // Layer 1: X @ W1 (320->128) for both heads
    gemm_silu(sX, K1P, 10, Ws1,  bs1,  8, H1s,  128, lane, wave);
    gemm_silu(sX, K1P, 10, Wab1, bab1, 8, H1ab, 128, lane, wave);
    __syncthreads();
    // Layer 2: 128->128
    gemm_silu(H1s,  128, 4, Ws2,  bs2,  8, H2s,  128, lane, wave);
    gemm_silu(H1ab, 128, 4, Wab2, bab2, 8, H2ab, 128, lane, wave);
    __syncthreads();
    // Layer 3: accumulate in registers (outputs overlap source LDS regions)
    v8f acc3s[2];
    v8f acc3ab[4];
#pragma unroll
    for (int t = 0; t < 2; ++t) {
        int tile = wave + t * 8;
        acc3s[t] = wmma_tile(H2s, 128, 4, tile >> 3, Ws3, 8, tile & 7, lane);
    }
#pragma unroll
    for (int t = 0; t < 4; ++t) {
        int tile = wave + t * 8;
        acc3ab[t] = wmma_tile(H2ab, 128, 4, tile >> 4, Wab3, 16, tile & 15, lane);
    }
    __syncthreads();
#pragma unroll
    for (int t = 0; t < 2; ++t) {
        int tile = wave + t * 8;
        int nt = tile & 7;
        int n = nt * 16 + (lane & 15);
        int mb = (tile >> 3) * 16 + ((lane & 16) ? 8 : 0);
        float bv = bs3[n];
#pragma unroll
        for (int v = 0; v < 8; ++v) outS[(mb + v) * 128 + n] = acc3s[t][v] + bv;
    }
#pragma unroll
    for (int t = 0; t < 4; ++t) {
        int tile = wave + t * 8;
        int nt = tile & 15;
        int n = nt * 16 + (lane & 15);
        int mb = (tile >> 4) * 16 + ((lane & 16) ? 8 : 0);
        float bv = bab3[n];
#pragma unroll
        for (int v = 0; v < 8; ++v) outAB[(mb + v) * 256 + n] = acc3ab[t][v] + bv;
    }
    __syncthreads();

    // Masked reductions.
    if (tid < 128) {
        float sum = 0.f;
        for (int j = 0; j < 32; ++j) sum += outS[j * 128 + tid] * mskS[j];
        atomicAdd(&M_s[bi * CSc + tid], sum);
    }
    for (int it = tid; it < 384; it += 256) {
        int d = it >> 7;
        int c = it & 127;
        float sum = 0.f;
        for (int j = 0; j < 32; ++j) {
            float mk = mskS[j];
            float al = outAB[j * 256 + c] * mk;
            float bt = outAB[j * 256 + 128 + c] * mk;
            sum += al * h_v[((b * Nn + j0 + j) * 3 + d) * CVc + c] +
                   bt * rhat[(pbase + j) * 3 + d];
        }
        atomicAdd(&M_v[(bi * 3 + d) * CVc + c], sum);
    }
}

// ---------------------------------------------------------------------------
// Node update: norms, U_s + gate MLPs (259->128->128->128), residual updates.
// One 128-thread block per node.
// ---------------------------------------------------------------------------
__global__ void node_update_kernel(
    const float* __restrict__ M_s, const float* __restrict__ M_v,
    const float* uW1, const float* uB1, const float* uW2, const float* uB2,
    const float* uW3, const float* uB3,
    const float* gW1, const float* gB1, const float* gW2, const float* gB2,
    const float* gW3, const float* gB3,
    float* __restrict__ h_s, float* __restrict__ h_v) {
    int node = blockIdx.x;
    int t = threadIdx.x;
    __shared__ float upd[259], u1[128], u2[128], g1[128], g2[128], red[128];

    float pnv = 0.f, pnm = 0.f, pdot = 0.f;
    for (int k = t; k < 384; k += 128) {
        float a = h_v[node * 384 + k];
        float m = M_v[node * 384 + k];
        pnv += a * a; pnm += m * m; pdot += a * m;
    }
    float sums[3];
    float vals[3] = {pnv, pnm, pdot};
#pragma unroll
    for (int s = 0; s < 3; ++s) {
        red[t] = vals[s];
        __syncthreads();
        for (int w = 64; w > 0; w >>= 1) {
            if (t < w) red[t] += red[t + w];
            __syncthreads();
        }
        sums[s] = red[0];
        __syncthreads();
    }
    upd[t] = h_s[node * CSc + t];
    upd[128 + t] = M_s[node * CSc + t];
    if (t == 0) {
        upd[256] = sqrtf(fmaxf(sums[0], EPSf));
        upd[257] = sqrtf(fmaxf(sums[1], EPSf));
        upd[258] = sums[2];
    }
    __syncthreads();
    {
        float au = uB1[t], ag = gB1[t];
        for (int k = 0; k < 259; ++k) {
            float x = upd[k];
            au += x * uW1[k * 128 + t];
            ag += x * gW1[k * 128 + t];
        }
        u1[t] = silu_f(au);
        g1[t] = silu_f(ag);
    }
    __syncthreads();
    {
        float au = uB2[t], ag = gB2[t];
        for (int k = 0; k < 128; ++k) {
            au += u1[k] * uW2[k * 128 + t];
            ag += g1[k] * gW2[k * 128 + t];
        }
        u2[t] = silu_f(au);
        g2[t] = silu_f(ag);
    }
    __syncthreads();
    {
        float au = uB3[t], ag = gB3[t];
        for (int k = 0; k < 128; ++k) {
            au += u2[k] * uW3[k * 128 + t];
            ag += g2[k] * gW3[k * 128 + t];
        }
        h_s[node * CSc + t] += au;
        float gv = tanhf(ag);
#pragma unroll
        for (int d = 0; d < 3; ++d) {
            int idx = (node * 3 + d) * CVc + t;
            h_v[idx] += gv * M_v[idx];
        }
    }
}

// ---------------------------------------------------------------------------
// Output: concat(h_s [1024,128], h_v [1024,384]) -> [1024,512] f32
// ---------------------------------------------------------------------------
__global__ void writeout_kernel(const float* __restrict__ h_s,
                                const float* __restrict__ h_v,
                                float* __restrict__ out) {
    int idx = blockIdx.x * blockDim.x + threadIdx.x;
    if (idx >= NODES * 512) return;
    int node = idx >> 9;
    int c = idx & 511;
    out[idx] = (c < 128) ? h_s[node * CSc + c] : h_v[node * 384 + (c - 128)];
}

// ---------------------------------------------------------------------------
// Host launcher
// ---------------------------------------------------------------------------
extern "C" void kernel_launch(void* const* d_in, const int* in_sizes, int n_in,
                              void* d_out, int out_size, void* d_ws, size_t ws_size,
                              hipStream_t stream) {
    (void)in_sizes; (void)n_in; (void)out_size; (void)ws_size;
    const float* qm     = (const float*)d_in[0];
    const float* mmc    = (const float*)d_in[1];
    const float* qeff   = (const float*)d_in[2];
    const int*   realmm = (const int*)d_in[3];
    const int*   atype  = (const int*)d_in[4];
    int p = 5;
    auto F = [&](void) { return (const float*)d_in[p++]; };
    const float* embed = F();
    const float *pW[3], *pB[3];
    for (int l = 0; l < 3; ++l) { pW[l] = F(); pB[l] = F(); }
    const float* spW = F();
    const float* spB = F();
    const float *eW[3], *eB[3];
    for (int l = 0; l < 3; ++l) { eW[l] = F(); eB[l] = F(); }
    const float *sW[3][3], *sB[3][3], *abW[3][3], *abB[3][3];
    const float *uW[3][3], *uB[3][3], *gW[3][3], *gB[3][3];
    for (int L = 0; L < 3; ++L) {
        for (int l = 0; l < 3; ++l) { sW[L][l] = F();  sB[L][l] = F();  }
        for (int l = 0; l < 3; ++l) { abW[L][l] = F(); abB[L][l] = F(); }
        for (int l = 0; l < 3; ++l) { uW[L][l] = F();  uB[L][l] = F();  }
        for (int l = 0; l < 3; ++l) { gW[L][l] = F();  gB[L][l] = F();  }
    }

    // Workspace carve (256B aligned)
    char* ws = (char*)d_ws;
    size_t off = 0;
    auto alloc = [&](size_t bytes) -> char* {
        char* r = ws + off;
        off += (bytes + 255) & ~(size_t)255;
        return r;
    };
    float* phi   = (float*)alloc(NODES * 4);
    float* evec  = (float*)alloc(NODES * 3 * 4);
    float* enorm = (float*)alloc(NODES * 4);
    float* h_s   = (float*)alloc(NODES * CSc * 4);
    float* h_v   = (float*)alloc(NODES * 384 * 4);
    float* rhat  = (float*)alloc(PAIRS * 3 * 4);
    _Float16* eij = (_Float16*)alloc((size_t)PAIRS * ERr * 2);
    float* maskf = (float*)alloc(PAIRS * 4);
    float* c1 = (float*)alloc(PAIRS * 4);
    float* c2 = (float*)alloc(PAIRS * 4);
    float* c3 = (float*)alloc(PAIRS * 4);
    float* M_s = (float*)alloc(NODES * CSc * 4);
    float* M_v = (float*)alloc(NODES * 384 * 4);
    _Float16 *wp_s1[3], *wp_s2[3], *wp_s3[3], *wp_ab1[3], *wp_ab2[3], *wp_ab3[3];
    for (int L = 0; L < 3; ++L) {
        wp_s1[L]  = (_Float16*)alloc(320 * 128 * 2);
        wp_s2[L]  = (_Float16*)alloc(128 * 128 * 2);
        wp_s3[L]  = (_Float16*)alloc(128 * 128 * 2);
        wp_ab1[L] = (_Float16*)alloc(320 * 128 * 2);
        wp_ab2[L] = (_Float16*)alloc(128 * 128 * 2);
        wp_ab3[L] = (_Float16*)alloc(128 * 256 * 2);
    }

    // Pack fp32 weights -> f16 WMMA B-fragment layout
    for (int L = 0; L < 3; ++L) {
        pack_w_kernel<<<128, 256, 0, stream>>>(sW[L][0],  wp_s1[L],  291, 128, 320);
        pack_w_kernel<<<128, 256, 0, stream>>>(sW[L][1],  wp_s2[L],  128, 128, 128);
        pack_w_kernel<<<128, 256, 0, stream>>>(sW[L][2],  wp_s3[L],  128, 128, 128);
        pack_w_kernel<<<128, 256, 0, stream>>>(abW[L][0], wp_ab1[L], 291, 128, 320);
        pack_w_kernel<<<128, 256, 0, stream>>>(abW[L][1], wp_ab2[L], 128, 128, 128);
        pack_w_kernel<<<128, 256, 0, stream>>>(abW[L][2], wp_ab3[L], 128, 256, 128);
    }

    esp_kernel<<<NODES, 256, 0, stream>>>(qm, mmc, qeff, realmm, phi, evec, enorm);
    init_nodes_kernel<<<NODES, 128, 0, stream>>>(
        phi, enorm, evec, atype, embed, pW[0], pB[0], pW[1], pB[1], pW[2], pB[2],
        spW, spB, eW[0], eB[0], eW[1], eB[1], eW[2], eB[2], h_s, h_v);
    geometry_kernel<<<PAIRS / 256, 256, 0, stream>>>(qm, rhat, eij, maskf);

    for (int L = 0; L < 3; ++L) {
        pair_scalars_kernel<<<NODES, 64, 0, stream>>>(h_v, rhat, c1, c2, c3);
        hipMemsetAsync(M_s, 0, NODES * CSc * 4, stream);
        hipMemsetAsync(M_v, 0, NODES * 384 * 4, stream);
        pair_mlp_kernel<<<NODES * 2, 256, 0, stream>>>(
            h_s, eij, c1, c2, c3, maskf, rhat, h_v,
            wp_s1[L], wp_s2[L], wp_s3[L], wp_ab1[L], wp_ab2[L], wp_ab3[L],
            sB[L][0], sB[L][1], sB[L][2], abB[L][0], abB[L][1], abB[L][2],
            M_s, M_v);
        node_update_kernel<<<NODES, 128, 0, stream>>>(
            M_s, M_v, uW[L][0], uB[L][0], uW[L][1], uB[L][1], uW[L][2], uB[L][2],
            gW[L][0], gB[L][0], gW[L][1], gB[L][1], gW[L][2], gB[L][2], h_s, h_v);
    }
    writeout_kernel<<<(NODES * 512 + 255) / 256, 256, 0, stream>>>(h_s, h_v,
                                                                   (float*)d_out);
}
